// LoopGNN_25314537243266
// MI455X (gfx1250) — matrix operand order repaired and checked by reading, not compiled, not run on hardware
//
#include <hip/hip_runtime.h>
#include <hip/hip_bf16.h>

// ---------------- vector types ----------------
typedef __attribute__((ext_vector_type(16))) _Float16 v16h_t;
typedef __attribute__((ext_vector_type(8)))  _Float16 v8h_t;
typedef __attribute__((ext_vector_type(4)))  _Float16 v4h_t;
typedef __attribute__((ext_vector_type(2)))  _Float16 v2h_t;
typedef __attribute__((ext_vector_type(8)))  float    v8f_t;
typedef __attribute__((ext_vector_type(4)))  float    v4f_t;

#define LDSS 40   // padded LDS row stride (halfs); 80B rows stay 16B aligned

// ---- fragment loaders (CDNA5 16-bit WMMA layouts, wave32) ----
// A 16x32: lanes 0-15 hold M=lane, K=0..7 & 16..23; lanes 16-31 hold K=8..15 & 24..31
__device__ __forceinline__ v16h_t load_a_frag(const _Float16* As, int mt, int lane) {
    int m  = mt * 16 + (lane & 15);
    int kb = (lane < 16) ? 0 : 8;
    const v8h_t* p = (const v8h_t*)(As + m * LDSS + kb);
    v8h_t lo = p[0];   // K = kb .. kb+7
    v8h_t hi = p[2];   // K = kb+16 .. kb+23
    v16h_t r;
#pragma unroll
    for (int j = 0; j < 8; ++j) { r[j] = lo[j]; r[j + 8] = hi[j]; }
    return r;
}
// B 32x16 (staged transposed in LDS as Bs[n][k]): lane n=lane&15,
// lanes 0-15 hold K=0..7 & 16..23; lanes 16-31 hold K=8..15 & 24..31
__device__ __forceinline__ v16h_t load_b_frag(const _Float16* Bs, int nt, int lane) {
    int n  = nt * 16 + (lane & 15);
    int kb = (lane < 16) ? 0 : 8;
    const v8h_t* p = (const v8h_t*)(Bs + n * LDSS + kb);
    v8h_t lo = p[0], hi = p[2];
    v16h_t r;
#pragma unroll
    for (int j = 0; j < 8; ++j) { r[j] = lo[j]; r[j + 8] = hi[j]; }
    return r;
}

__device__ __forceinline__ float nclean(float v, int flags) {
    if (flags & 2) v = (v == v) ? v : 0.f;   // nan_to_num (branchless)
    return v;
}

// flags: 1=relu, 2=nan_to_num A, 4=permute B rows (vlad->hw), 8=NT A loads,
//        16=NT B loads, 32=NT C stores
__device__ __forceinline__ void stage_a(const float* __restrict__ A, _Float16* As,
                                        int tid, int m0, int k0, int M, int K,
                                        int lda, int flags) {
    int row = tid >> 2, col = (tid & 3) * 8;
    int gm = m0 + row;
    v8h_t h;
    if (m0 + 64 <= M && k0 + 32 <= K) {
        const float* ap = A + (long long)gm * lda + k0 + col;
        v4f_t u, w;
        if (flags & 8) {
            u = __builtin_nontemporal_load((const v4f_t*)ap);
            w = __builtin_nontemporal_load((const v4f_t*)ap + 1);
        } else {
            u = *(const v4f_t*)ap;
            w = *((const v4f_t*)ap + 1);
        }
        if (k0 + 96 <= K) __builtin_prefetch(ap + 64, 0, 1);  // global_prefetch_b8, 2 chunks ahead
#pragma unroll
        for (int j = 0; j < 4; ++j) {
            h[j]     = (_Float16)nclean(u[j], flags);
            h[j + 4] = (_Float16)nclean(w[j], flags);
        }
    } else {
        int gmc = gm < M ? gm : (M - 1);
        const float* ap = A + (long long)gmc * lda;
#pragma unroll
        for (int j = 0; j < 8; ++j) {
            int gk = k0 + col + j;
            int gkc = gk < K ? gk : (K - 1);
            float v = ap[gkc];
            v = (gm < M && gk < K) ? nclean(v, flags) : 0.f;
            h[j] = (_Float16)v;
        }
    }
    *(v8h_t*)(As + row * LDSS + col) = h;   // one ds_store_b128
}

__device__ __forceinline__ long long b_row(int gk, int flags) {
    return (flags & 4) ? ((long long)(gk & 255) * 64 + (gk >> 8))  // vladT -> hw perm
                       : (long long)gk;
}

// transpose-stage B: each thread covers 2(k) x 4(n); packs k-pairs -> ds_store_b32
__device__ __forceinline__ void stage_b(const float* __restrict__ B, _Float16* Bs,
                                        int tid, int n0, int k0, int N, int K,
                                        int ldb, int flags) {
    int kk = (tid >> 4) * 2;          // 0,2,..,30
    int nb = (tid & 15) * 4;          // 0,4,..,60
    int gk0 = k0 + kk, gk1 = gk0 + 1;
    if (n0 + 64 <= N && k0 + 32 <= K) {
        const v4f_t* q0 = (const v4f_t*)(B + b_row(gk0, flags) * ldb + n0 + nb);
        const v4f_t* q1 = (const v4f_t*)(B + b_row(gk1, flags) * ldb + n0 + nb);
        v4f_t u, w;
        if (flags & 16) { u = __builtin_nontemporal_load(q0); w = __builtin_nontemporal_load(q1); }
        else            { u = *q0; w = *q1; }
#pragma unroll
        for (int j = 0; j < 4; ++j) {
            v2h_t pk; pk[0] = (_Float16)u[j]; pk[1] = (_Float16)w[j];
            *(v2h_t*)(Bs + (nb + j) * LDSS + kk) = pk;
        }
    } else {
#pragma unroll
        for (int jj = 0; jj < 2; ++jj) {
            int gk = gk0 + jj;
            int gkc = gk < K ? gk : (K - 1);
            const float* bp = B + b_row(gkc, flags) * ldb;
#pragma unroll
            for (int j = 0; j < 4; ++j) {
                int gn = n0 + nb + j;
                int gnc = gn < N ? gn : (N - 1);
                float v = bp[gnc];
                v = (gk < K && gn < N) ? v : 0.f;
                Bs[(nb + j) * LDSS + kk + jj] = (_Float16)v;
            }
        }
    }
}

// ---------------- generic tiled WMMA GEMM (double-buffered LDS) ----------------
// C[M,N] = act( A[M,K] @ B[K,N] + bias )    (A,B,C fp32; compute f16 WMMA)
__global__ __launch_bounds__(256) void gemm_wmma_kernel(
    const float* __restrict__ A, const float* __restrict__ B,
    const float* __restrict__ bias, float* __restrict__ C,
    int M, int N, int K, int lda, int ldb, int ldc, int flags)
{
    __shared__ __align__(16) _Float16 As[2][64 * LDSS];
    __shared__ __align__(16) _Float16 Bs[2][64 * LDSS];  // transposed: Bs[n][k]
    const int tid = threadIdx.x, wave = tid >> 5, lane = tid & 31;
    const int m0 = blockIdx.x * 64, n0 = blockIdx.y * 64;
    const int mt = wave & 3, ntb = (wave >> 2) * 2;
    v8f_t c0 = {}; v8f_t c1 = {};

    const int nchunks = (K + 31) >> 5;
    stage_a(A, As[0], tid, m0, 0, M, K, lda, flags);
    stage_b(B, Bs[0], tid, n0, 0, N, K, ldb, flags);
    __syncthreads();
    for (int ci = 0; ci < nchunks; ++ci) {
        const int cur = ci & 1;
        if (ci + 1 < nchunks) {  // stage next chunk while WMMA consumes current
            stage_a(A, As[cur ^ 1], tid, m0, (ci + 1) * 32, M, K, lda, flags);
            stage_b(B, Bs[cur ^ 1], tid, n0, (ci + 1) * 32, N, K, ldb, flags);
        }
        v16h_t af = load_a_frag(As[cur], mt, lane);
        v16h_t b0 = load_b_frag(Bs[cur], ntb + 0, lane);
        c0 = __builtin_amdgcn_wmma_f32_16x16x32_f16(false, af, false, b0, (short)0, c0, false, false);
        v16h_t b1 = load_b_frag(Bs[cur], ntb + 1, lane);
        c1 = __builtin_amdgcn_wmma_f32_16x16x32_f16(false, af, false, b1, (short)0, c1, false, false);
        __syncthreads();
    }
    // epilogue: C layout VGPR r -> M = r + (lane<16?0:8), N = lane&15
#pragma unroll
    for (int t = 0; t < 2; ++t) {
        v8f_t c = t ? c1 : c0;
        int ncol = n0 + (ntb + t) * 16 + (lane & 15);
        if (ncol >= N) continue;
        int mbase = m0 + mt * 16 + ((lane < 16) ? 0 : 8);
#pragma unroll
        for (int r = 0; r < 8; ++r) {
            int gm = mbase + r;
            if (gm >= M) continue;
            float v = c[r];
            if (bias) v += bias[ncol];
            if (flags & 1) v = fmaxf(v, 0.f);
            float* cp = C + (long long)gm * ldc + ncol;
            if (flags & 32) __builtin_nontemporal_store(v, cp);
            else *cp = v;
        }
    }
}

// ---------------- batched NetVLAD residual GEMM (double-buffered) ----------------
// per node b: vladT[b, c(64), d(256)] = sum_n act[b,n,c] * X[b,n,d] - asum[b,c]*cw2[d,c]
__global__ __launch_bounds__(256) void vlad_wmma_kernel(
    const _Float16* __restrict__ act,   // [1e6, 64]
    const float* __restrict__ X,        // [1e6, 256]
    const float* __restrict__ asum,     // [2000, 64]
    const float* __restrict__ cw2,      // [256, 64]  (d,k)
    float* __restrict__ vladT)          // [2000, 64, 256]
{
    __shared__ __align__(16) _Float16 As[2][64 * LDSS];  // [cluster][kk]
    __shared__ __align__(16) _Float16 Bs[2][64 * LDSS];  // transposed: [d][kk]
    const int tid = threadIdx.x, wave = tid >> 5, lane = tid & 31;
    const int b = blockIdx.y;
    const int n0 = blockIdx.x * 64;                      // d-tile
    const int mt = wave & 3, ntb = (wave >> 2) * 2;
    v8f_t c0 = {}; v8f_t c1 = {};
    const long long rowbase = (long long)b * 500;

    // stage one 32-keypoint chunk; 2(k) x 4(col) per thread, k-pairs packed
    auto stage = [&](int buf, int k0) {
        int kk = (tid >> 4) * 2, cb = (tid & 15) * 4;
        bool ok0 = (k0 + kk)     < 500;
        bool ok1 = (k0 + kk + 1) < 500;
        long long r0 = ok0 ? (rowbase + k0 + kk)     : (rowbase + 499);
        long long r1 = ok1 ? (rowbase + k0 + kk + 1) : (rowbase + 499);
        { // A from act (f16): transpose into As[cluster][kk]
            v4h_t a0 = *(const v4h_t*)(act + r0 * 64 + cb);
            v4h_t a1 = *(const v4h_t*)(act + r1 * 64 + cb);
#pragma unroll
            for (int j = 0; j < 4; ++j) {
                v2h_t pk;
                pk[0] = ok0 ? a0[j] : (_Float16)0.f;
                pk[1] = ok1 ? a1[j] : (_Float16)0.f;
                *(v2h_t*)(As[buf] + (cb + j) * LDSS + kk) = pk;
            }
        }
        { // B from X (f32, single-pass stream -> NT), nan_to_num
            v4f_t u = __builtin_nontemporal_load((const v4f_t*)(X + r0 * 256 + n0 + cb));
            v4f_t w = __builtin_nontemporal_load((const v4f_t*)(X + r1 * 256 + n0 + cb));
#pragma unroll
            for (int j = 0; j < 4; ++j) {
                float x0 = u[j], x1 = w[j];
                x0 = (x0 == x0) ? x0 : 0.f;  x1 = (x1 == x1) ? x1 : 0.f;
                v2h_t pk;
                pk[0] = ok0 ? (_Float16)x0 : (_Float16)0.f;
                pk[1] = ok1 ? (_Float16)x1 : (_Float16)0.f;
                *(v2h_t*)(Bs[buf] + (cb + j) * LDSS + kk) = pk;
            }
        }
    };

    stage(0, 0);
    __syncthreads();
#pragma unroll 1
    for (int ci = 0; ci < 16; ++ci) {
        const int cur = ci & 1;
        if (ci + 1 < 16) stage(cur ^ 1, (ci + 1) * 32);
        v16h_t af = load_a_frag(As[cur], mt, lane);
        v16h_t b0 = load_b_frag(Bs[cur], ntb + 0, lane);
        c0 = __builtin_amdgcn_wmma_f32_16x16x32_f16(false, af, false, b0, (short)0, c0, false, false);
        v16h_t b1 = load_b_frag(Bs[cur], ntb + 1, lane);
        c1 = __builtin_amdgcn_wmma_f32_16x16x32_f16(false, af, false, b1, (short)0, c1, false, false);
        __syncthreads();
    }
#pragma unroll
    for (int t = 0; t < 2; ++t) {
        v8f_t c = t ? c1 : c0;
        int gn = n0 + (ntb + t) * 16 + (lane & 15);           // d
        int mbase = mt * 16 + ((lane < 16) ? 0 : 8);          // cluster
#pragma unroll
        for (int r = 0; r < 8; ++r) {
            int gm = mbase + r;
            float v = c[r] - asum[b * 64 + gm] * cw2[gn * 64 + gm];
            vladT[((long long)b * 64 + gm) * 256 + gn] = v;
        }
    }
}

// ---------------- VALU helper kernels ----------------
__global__ void fill_f32(float* p, float v, long long n) {
    long long i = (long long)blockIdx.x * blockDim.x + threadIdx.x;
    if (i < n) p[i] = v;
}

// per-row softmax over 64 logits; writes act f16 and accumulates act_sum[b,k]
__global__ void softmax_rows(const float* __restrict__ logits, _Float16* __restrict__ act,
                             float* __restrict__ asum, int rows) {
    long long t = (long long)blockIdx.x * blockDim.x + threadIdx.x;
    int row = (int)(t >> 5), lane = (int)(t & 31);
    if (row >= rows) return;
    const float* lp = logits + (long long)row * 64;
    float v0 = __builtin_nontemporal_load(lp + lane);        // single-pass 256MB stream
    float v1 = __builtin_nontemporal_load(lp + lane + 32);
    float m = fmaxf(v0, v1);
#pragma unroll
    for (int off = 16; off; off >>= 1) m = fmaxf(m, __shfl_xor(m, off, 32));
    float e0 = __expf(v0 - m), e1 = __expf(v1 - m);
    float s = e0 + e1;
#pragma unroll
    for (int off = 16; off; off >>= 1) s += __shfl_xor(s, off, 32);
    float inv = 1.f / s;
    float a0 = e0 * inv, a1 = e1 * inv;
    act[(long long)row * 64 + lane] = (_Float16)a0;
    act[(long long)row * 64 + lane + 32] = (_Float16)a1;
    int b = row / 500;
    atomicAdd(&asum[b * 64 + lane], a0);
    atomicAdd(&asum[b * 64 + lane + 32], a1);
}

// intra-norm over d (256) per (b,cluster) row; accumulate total sumsq per b
__global__ void intra_norm(float* __restrict__ vladT, float* __restrict__ total, int rows) {
    long long t = (long long)blockIdx.x * blockDim.x + threadIdx.x;
    int row = (int)(t >> 5), lane = (int)(t & 31);
    if (row >= rows) return;
    float* p = vladT + (long long)row * 256;
    float ss = 0.f;
    for (int j = lane; j < 256; j += 32) { float v = p[j]; ss += v * v; }
#pragma unroll
    for (int off = 16; off; off >>= 1) ss += __shfl_xor(ss, off, 32);
    float rn = 1.f / (sqrtf(ss) + 1e-12f);
    for (int j = lane; j < 256; j += 32) p[j] *= rn;
    if (lane == 0) atomicAdd(&total[row >> 6], ss * rn * rn);
}

__global__ void global_scale(float* __restrict__ vladT, const float* __restrict__ total, long long n) {
    long long i = (long long)blockIdx.x * blockDim.x + threadIdx.x;
    if (i >= n) return;
    int b = (int)(i >> 14);
    vladT[i] *= 1.f / (sqrtf(total[b]) + 1e-12f);
}

// per-node attention dot products (wave per node)
__global__ void gat_scores(const float* __restrict__ h, const float* __restrict__ a_s,
                           const float* __restrict__ a_d, float* __restrict__ ssrc,
                           float* __restrict__ sdst, int n) {
    long long t = (long long)blockIdx.x * blockDim.x + threadIdx.x;
    int node = (int)(t >> 5), lane = (int)(t & 31);
    if (node >= n) return;
    float s = 0.f, d = 0.f;
    for (int j = lane; j < 128; j += 32) {
        float v = h[(long long)node * 128 + j];
        s += v * a_s[j]; d += v * a_d[j];
    }
#pragma unroll
    for (int off = 16; off; off >>= 1) { s += __shfl_xor(s, off, 32); d += __shfl_xor(d, off, 32); }
    if (lane == 0) { ssrc[node] = s; sdst[node] = d; }
}

__device__ __forceinline__ void atomicMaxF(float* addr, float val) {
    int cur = __float_as_int(*addr);
    while (val > __int_as_float(cur)) {
        int old = cur;
        cur = atomicCAS((int*)addr, old, __float_as_int(val));
        if (cur == old) break;
    }
}

__global__ void gat_edge1(const int* __restrict__ ei, const float* __restrict__ ssrc,
                          const float* __restrict__ sdst, float* __restrict__ ebuf,
                          float* __restrict__ nmax, int nE, int nN) {
    int i = blockIdx.x * blockDim.x + threadIdx.x;
    if (i >= nE + nN) return;
    int s, d;
    if (i < nE) { s = ei[i]; d = ei[nE + i]; } else { s = d = i - nE; }
    float v = ssrc[s] + sdst[d];
    v = v > 0.f ? v : 0.2f * v;             // leaky_relu(0.2)
    ebuf[i] = v;
    atomicMaxF(&nmax[d], v);
}

__global__ void gat_edge2(const int* __restrict__ ei, const float* __restrict__ ebuf,
                          const float* __restrict__ nmax, float* __restrict__ exbuf,
                          float* __restrict__ nsum, int nE, int nN) {
    int i = blockIdx.x * blockDim.x + threadIdx.x;
    if (i >= nE + nN) return;
    int d = (i < nE) ? ei[nE + i] : (i - nE);
    float ex = __expf(ebuf[i] - nmax[d]);
    exbuf[i] = ex;
    atomicAdd(&nsum[d], ex);
}

__global__ void gat_aggregate(const int* __restrict__ ei, const float* __restrict__ h,
                              const float* __restrict__ exbuf, const float* __restrict__ nsum,
                              float* __restrict__ xagg, int nE, int nN) {
    long long t = (long long)blockIdx.x * blockDim.x + threadIdx.x;
    int edge = (int)(t >> 5), lane = (int)(t & 31);
    if (edge >= nE + nN) return;
    int s, d;
    if (edge < nE) { s = ei[edge]; d = ei[nE + edge]; } else { s = d = edge - nE; }
    float alpha = exbuf[edge] / nsum[d];
    for (int j = lane; j < 128; j += 32)
        atomicAdd(&xagg[(long long)d * 128 + j], h[(long long)s * 128 + j] * alpha);
}

__global__ void gat_finalize(const float* __restrict__ xagg, const float* __restrict__ bias,
                             float* __restrict__ xout, int n) {
    int i = blockIdx.x * blockDim.x + threadIdx.x;
    if (i >= n) return;
    float v = xagg[i] + bias[i & 127];
    xout[i] = v > 0.f ? v : (__expf(v) - 1.f);  // elu
}

__global__ void edge_encoder(const float* __restrict__ ea,
                             const float* __restrict__ w1, const float* __restrict__ b1,
                             const float* __restrict__ w2, const float* __restrict__ b2,
                             const float* __restrict__ w3, const float* __restrict__ b3,
                             float* __restrict__ ef, int nE) {
    int e = blockIdx.x * blockDim.x + threadIdx.x;
    if (e >= nE) return;
    float a = ea[e];
    float h8[8], h16[16];
#pragma unroll
    for (int j = 0; j < 8; ++j) h8[j] = fmaxf(a * w1[j] + b1[j], 0.f);
#pragma unroll
    for (int j = 0; j < 16; ++j) {
        float s = b2[j];
#pragma unroll
        for (int k = 0; k < 8; ++k) s += h8[k] * w2[k * 16 + j];
        h16[j] = fmaxf(s, 0.f);
    }
#pragma unroll
    for (int j = 0; j < 32; ++j) {
        float s = b3[j];
#pragma unroll
        for (int k = 0; k < 16; ++k) s += h16[k] * w3[k * 32 + j];
        ef[(long long)e * 32 + j] = s;
    }
}

__global__ void edge_classifier(const int* __restrict__ ei, const float* __restrict__ x,
                                const float* __restrict__ ef,
                                const float* __restrict__ dpw, const float* __restrict__ dpb,
                                const float* __restrict__ w1, const float* __restrict__ b1,
                                const float* __restrict__ w2, const float* __restrict__ b2,
                                float* __restrict__ out, int nE) {
    int e = blockIdx.x * blockDim.x + threadIdx.x;
    if (e >= nE) return;
    int s = ei[e], d = ei[nE + e];
    float acc[32];
#pragma unroll
    for (int j = 0; j < 32; ++j) acc[j] = dpb[j];
    for (int k = 0; k < 288; ++k) {
        float f = (k < 128) ? x[(long long)s * 128 + k]
                 : (k < 256) ? x[(long long)d * 128 + (k - 128)]
                             : ef[(long long)e * 32 + (k - 256)];
        const float* wr = dpw + k * 32;
#pragma unroll
        for (int j = 0; j < 32; ++j) acc[j] += f * wr[j];
    }
    float h1[16];
#pragma unroll
    for (int j = 0; j < 16; ++j) {
        float t = b1[j];
#pragma unroll
        for (int k = 0; k < 32; ++k) t += acc[k] * w1[k * 16 + j];
        h1[j] = fmaxf(t, 0.f);
    }
    float o = b2[0];
#pragma unroll
    for (int k = 0; k < 16; ++k) o += h1[k] * w2[k];
    out[e] = 1.f / (1.f + __expf(-o));
}

// ---------------- host-side orchestration ----------------
extern "C" void kernel_launch(void* const* d_in, const int* in_sizes, int n_in,
                              void* d_out, int out_size, void* d_ws, size_t ws_size,
                              hipStream_t stream) {
    (void)in_sizes; (void)n_in; (void)out_size; (void)ws_size;
    const float* node_feats = (const float*)d_in[0];
    const float* edge_attr  = (const float*)d_in[1];
    const float* nv_cw   = (const float*)d_in[2];
    const float* nv_cb   = (const float*)d_in[3];
    const float* nv_cw2  = (const float*)d_in[4];
    const float* nv_hw   = (const float*)d_in[5];
    const float* ne_w1   = (const float*)d_in[6];
    const float* ne_b1   = (const float*)d_in[7];
    const float* ne_w2   = (const float*)d_in[8];
    const float* ne_b2   = (const float*)d_in[9];
    const float* ee_w1   = (const float*)d_in[10];
    const float* ee_b1   = (const float*)d_in[11];
    const float* ee_w2   = (const float*)d_in[12];
    const float* ee_b2   = (const float*)d_in[13];
    const float* ee_w3   = (const float*)d_in[14];
    const float* ee_b3   = (const float*)d_in[15];
    const float* gat_w   = (const float*)d_in[16];
    const float* gat_as  = (const float*)d_in[17];
    const float* gat_ad  = (const float*)d_in[18];
    const float* gat_b   = (const float*)d_in[19];
    const float* dp_w    = (const float*)d_in[20];
    const float* dp_b    = (const float*)d_in[21];
    const float* ec_w1   = (const float*)d_in[22];
    const float* ec_b1   = (const float*)d_in[23];
    const float* ec_w2   = (const float*)d_in[24];
    const float* ec_b2   = (const float*)d_in[25];
    const int*   edge_index = (const int*)d_in[26];

    const int ROWS = 1000000, D = 256, KC = 64, NN = 2000;
    const int HID = 1000, MID = 500, ND = 128, NE = 64000, E2 = NE + NN;

    char* ws = (char*)d_ws;
    size_t off = 0;
    auto alloc = [&](size_t bytes) -> char* {
        char* p = ws + off;
        off = (off + bytes + 255) & ~(size_t)255;
        return p;
    };
    float*    logits = (float*)alloc((size_t)ROWS * KC * 4);        // 256 MB
    _Float16* act    = (_Float16*)alloc((size_t)ROWS * KC * 2);     // 128 MB
    float*    asum   = (float*)alloc((size_t)NN * KC * 4);
    float*    vladT  = (float*)alloc((size_t)NN * 16384 * 4);       // 128 MB
    float*    total  = (float*)alloc((size_t)NN * 4);
    float*    hidden = (float*)alloc((size_t)NN * HID * 4);
    float*    x1     = (float*)alloc((size_t)NN * MID * 4);
    float*    xa     = (float*)alloc((size_t)NN * ND * 4);
    float*    xb     = (float*)alloc((size_t)NN * ND * 4);
    float*    hbuf   = (float*)alloc((size_t)NN * ND * 4);
    float*    ssrc   = (float*)alloc((size_t)NN * 4);
    float*    sdst   = (float*)alloc((size_t)NN * 4);
    float*    ebuf   = (float*)alloc((size_t)E2 * 4);
    float*    exbuf  = (float*)alloc((size_t)E2 * 4);
    float*    nmax   = (float*)alloc((size_t)NN * 4);
    float*    nsum   = (float*)alloc((size_t)NN * 4);
    float*    xagg   = (float*)alloc((size_t)NN * ND * 4);
    float*    ef     = (float*)alloc((size_t)NE * 32 * 4);

    // 1) NetVLAD soft-assignment logits: X @ cw + cb
    //    (nan_to_num on A, NT A loads, NT C stores: both streams exceed 192MB L2)
    fill_f32<<<(NN * KC + 255) / 256, 256, 0, stream>>>(asum, 0.f, (long long)NN * KC);
    fill_f32<<<(NN + 255) / 256, 256, 0, stream>>>(total, 0.f, NN);
    gemm_wmma_kernel<<<dim3(ROWS / 64, 1), 256, 0, stream>>>(
        node_feats, nv_cw, nv_cb, logits, ROWS, KC, D, D, KC, KC, /*flags=*/2 | 8 | 32);
    // 2) row softmax -> act (f16) + act_sum
    softmax_rows<<<(int)(((long long)ROWS * 32 + 255) / 256), 256, 0, stream>>>(logits, act, asum, ROWS);
    // 3) per-node residual GEMM, fused (-act_sum*cw2) epilogue
    vlad_wmma_kernel<<<dim3(4, NN), 256, 0, stream>>>(act, node_feats, asum, nv_cw2, vladT);
    // 4) intra-norm over D, then global L2
    intra_norm<<<(NN * KC * 32 + 255) / 256, 256, 0, stream>>>(vladT, total, NN * KC);
    global_scale<<<(int)(((long long)NN * 16384 + 255) / 256), 256, 0, stream>>>(
        vladT, total, (long long)NN * 16384);
    // 5) hidden GEMM: vlad @ hw (RT caching: vladT 128MB + hw 64MB fit in 192MB L2;
    //    B rows permuted to absorb [k,d] vs [d,k] flatten order)
    gemm_wmma_kernel<<<dim3((NN + 63) / 64, (HID + 63) / 64), 256, 0, stream>>>(
        vladT, nv_hw, nullptr, hidden, NN, HID, 16384, 16384, HID, HID, /*flags=*/4);
    // 6) node encoder MLP
    gemm_wmma_kernel<<<dim3((NN + 63) / 64, (MID + 63) / 64), 256, 0, stream>>>(
        hidden, ne_w1, ne_b1, x1, NN, MID, HID, HID, MID, MID, /*flags=*/1);
    gemm_wmma_kernel<<<dim3((NN + 63) / 64, (ND + 63) / 64), 256, 0, stream>>>(
        x1, ne_w2, ne_b2, xa, NN, ND, MID, MID, ND, ND, 0);
    // 7) edge encoder MLP
    edge_encoder<<<(NE + 255) / 256, 256, 0, stream>>>(
        edge_attr, ee_w1, ee_b1, ee_w2, ee_b2, ee_w3, ee_b3, ef, NE);
    // 8) three GAT layers (self-loops appended as edges NE..NE+NN-1)
    float* xcur = xa; float* xnext = xb;
    for (int l = 0; l < 3; ++l) {
        gemm_wmma_kernel<<<dim3((NN + 63) / 64, (ND + 63) / 64), 256, 0, stream>>>(
            xcur, gat_w + (size_t)l * ND * ND, nullptr, hbuf, NN, ND, ND, ND, ND, ND, 0);
        gat_scores<<<(NN * 32 + 255) / 256, 256, 0, stream>>>(
            hbuf, gat_as + l * ND, gat_ad + l * ND, ssrc, sdst, NN);
        fill_f32<<<(NN + 255) / 256, 256, 0, stream>>>(nmax, -1e30f, NN);
        fill_f32<<<(NN + 255) / 256, 256, 0, stream>>>(nsum, 0.f, NN);
        fill_f32<<<(NN * ND + 255) / 256, 256, 0, stream>>>(xagg, 0.f, (long long)NN * ND);
        gat_edge1<<<(E2 + 255) / 256, 256, 0, stream>>>(edge_index, ssrc, sdst, ebuf, nmax, NE, NN);
        gat_edge2<<<(E2 + 255) / 256, 256, 0, stream>>>(edge_index, ebuf, nmax, exbuf, nsum, NE, NN);
        gat_aggregate<<<(int)(((long long)E2 * 32 + 255) / 256), 256, 0, stream>>>(
            edge_index, hbuf, exbuf, nsum, xagg, NE, NN);
        gat_finalize<<<(NN * ND + 255) / 256, 256, 0, stream>>>(xagg, gat_b + l * ND, xnext, NN * ND);
        float* t = xcur; xcur = xnext; xnext = t;
    }
    // 9) edge classifier -> sigmoid output
    edge_classifier<<<(NE + 255) / 256, 256, 0, stream>>>(
        edge_index, xcur, ef, dp_w, dp_b, ec_w1, ec_b1, ec_w2, ec_b2, (float*)d_out, NE);
}
